// RelationalGraphLayer_48180943127293
// MI455X (gfx1250) — matrix operand order repaired and checked by reading, not compiled, not run on hardware
//
#include <hip/hip_runtime.h>
#include <cstdint>
#include <cstddef>

typedef __bf16 bf16_t;
typedef __attribute__((ext_vector_type(16))) __bf16 v16bf;
typedef __attribute__((ext_vector_type(8)))  __bf16 v8bf;
typedef __attribute__((ext_vector_type(8)))  float  v8f;

#define NN   10000
#define EE   160000
#define DIMD 512
#define RR   4

// ---------------------------------------------------------------- zero
__global__ void k_zero(float* __restrict__ p, size_t n) {
  size_t i = (size_t)blockIdx.x * blockDim.x + threadIdx.x;
  size_t stride = (size_t)gridDim.x * blockDim.x;
  for (; i < n; i += stride) p[i] = 0.0f;
}

// ------------------------------------------- weight convert + transpose
// dst[f*dstStride + colOff + k] = bf16(src[k*F + f])   (write-coalesced)
__global__ void k_transpose_bf16(const float* __restrict__ src, bf16_t* __restrict__ dst,
                                 int K, int F, int dstStride, int colOff) {
  size_t total = (size_t)K * F;
  size_t stride = (size_t)gridDim.x * blockDim.x;
  for (size_t idx = (size_t)blockIdx.x * blockDim.x + threadIdx.x; idx < total; idx += stride) {
    int f = (int)(idx / (size_t)K);
    int k = (int)(idx % (size_t)K);
    dst[(size_t)f * dstStride + colOff + k] = (bf16_t)src[(size_t)k * F + f];
  }
}

// ---------------------------------------------------------- edge scatter
// s[dst, rel, :] += x[src, :]  ;  cnt[dst, rel] += 1    (one wave per edge)
__global__ void k_edge_scatter(const int* __restrict__ edge_index,
                               const int* __restrict__ edge_type,
                               const float* __restrict__ x,
                               float* __restrict__ s, float* __restrict__ cnt) {
  int e = blockIdx.x * 8 + (threadIdx.x >> 5);
  int lane = threadIdx.x & 31;
  int src = edge_index[e];
  int dst = edge_index[EE + e];
  int r   = edge_type[e];
  const float* xs = x + (size_t)src * DIMD;
  float* sd = s + ((size_t)dst * RR + r) * DIMD;
  #pragma unroll
  for (int i = 0; i < DIMD / 32; ++i)
    unsafeAtomicAdd(&sd[lane + 32 * i], xs[lane + 32 * i]);
  if (lane == 0) unsafeAtomicAdd(&cnt[dst * RR + r], 1.0f);
}

// -------------------------------------------------------- row finalize
// A0[n,f]   = x[n,f] + self_b[f] + invdeg * sum_r cnt[n,r]*rel_b[r,f]
// Acat[n,:] = [ bf16(x[n,:]) | bf16(s[n,:,:] * invdeg) ]     (K = 2560)
__global__ void k_finalize(const float* __restrict__ x,
                           const float* __restrict__ self_b,
                           const float* __restrict__ rel_b,
                           const float* __restrict__ s,
                           const float* __restrict__ cnt,
                           bf16_t* __restrict__ Acat, float* __restrict__ A0) {
  int n = blockIdx.x;
  int tid = threadIdx.x;
  float c0 = cnt[n * RR + 0], c1 = cnt[n * RR + 1];
  float c2 = cnt[n * RR + 2], c3 = cnt[n * RR + 3];
  float inv = 1.0f / fmaxf(c0 + c1 + c2 + c3, 1.0f);
  const float* xr = x + (size_t)n * DIMD;
  bf16_t* arow = Acat + (size_t)n * (DIMD * (RR + 1));
  float* a0r = A0 + (size_t)n * DIMD;
  for (int f = tid; f < DIMD; f += blockDim.x) {
    float xv = xr[f];
    a0r[f] = xv + self_b[f] +
             inv * (c0 * rel_b[f] + c1 * rel_b[DIMD + f] +
                    c2 * rel_b[2 * DIMD + f] + c3 * rel_b[3 * DIMD + f]);
    arow[f] = (bf16_t)xv;
  }
  const float* srow = s + (size_t)n * (RR * DIMD);
  for (int i = tid; i < RR * DIMD; i += blockDim.x)
    arow[DIMD + i] = (bf16_t)(srow[i] * inv);
}

// ----------------------------------------------------------- WMMA GEMM
// C[16 x 512-per-y] = A[m0:m0+16, 0:K] (bf16, row-major, ld=K)
//                   @ Bt (bf16, [OUTW][K] row-major, i.e. B transposed)
// mode 0: +bias +addend, LayerNorm over 512 cols -> outF (+ optional outB bf16)
// mode 1: +bias, exact GELU -> outB bf16
__global__ __launch_bounds__(256) void k_gemm_wmma(
    const bf16_t* __restrict__ A, const bf16_t* __restrict__ Bt,
    int K, int OUTW,
    const float* __restrict__ bias, const float* __restrict__ addend,
    const float* __restrict__ lnw, const float* __restrict__ lnb,
    float* __restrict__ outF, bf16_t* __restrict__ outB, int mode) {
  __shared__ float Ct[16][DIMD + 4];

  const int tid  = threadIdx.x;
  const int wave = tid >> 5;
  const int lane = tid & 31;
  const int m0 = blockIdx.x * 16;
  const int colBase = blockIdx.y * 512 + wave * 64;

  // A fragment addressing (16-bit A 16x32 layout):
  //   lane holds row (lane&15); K-chunk [kbA, kbA+16)
  const int rowA = lane & 15;
  const int kbA  = (lane >> 4) << 4;
  // B fragment addressing (16-bit B 32x16 layout):
  //   lane holds col (lane&15); K-chunk [kbB, kbB+16)
  const int nB  = lane & 15;
  const int kbB = (lane >> 4) << 4;

  const bf16_t* arowp = A + (size_t)(m0 + rowA) * K;
  const bf16_t* bp0 = Bt + (size_t)(colBase +  0 + nB) * K + kbB;
  const bf16_t* bp1 = Bt + (size_t)(colBase + 16 + nB) * K + kbB;
  const bf16_t* bp2 = Bt + (size_t)(colBase + 32 + nB) * K + kbB;
  const bf16_t* bp3 = Bt + (size_t)(colBase + 48 + nB) * K + kbB;

  v8f c0 = {}, c1 = {}, c2 = {}, c3 = {};

  for (int k0 = 0; k0 < K; k0 += 32) {
    v8bf lo = *(const v8bf*)(arowp + k0 + kbA);
    v8bf hi = *(const v8bf*)(arowp + k0 + kbA + 8);
    v16bf a;
    #pragma unroll
    for (int i = 0; i < 8; ++i) { a[i] = lo[i]; a[8 + i] = hi[i]; }
    v16bf b0 = *(const v16bf*)(bp0 + k0);
    v16bf b1 = *(const v16bf*)(bp1 + k0);
    v16bf b2 = *(const v16bf*)(bp2 + k0);
    v16bf b3 = *(const v16bf*)(bp3 + k0);
    c0 = __builtin_amdgcn_wmma_f32_16x16x32_bf16(false, a, false, b0, (short)0, c0, false, false);
    c1 = __builtin_amdgcn_wmma_f32_16x16x32_bf16(false, a, false, b1, (short)0, c1, false, false);
    c2 = __builtin_amdgcn_wmma_f32_16x16x32_bf16(false, a, false, b2, (short)0, c2, false, false);
    c3 = __builtin_amdgcn_wmma_f32_16x16x32_bf16(false, a, false, b3, (short)0, c3, false, false);
  }

  // C/D layout: lane<16 -> col=lane, rows = vgpr j; lane>=16 -> col=lane-16, rows = j+8
  const int hi16 = lane >> 4;

  if (mode == 1) {  // ------- GELU epilogue, elementwise -------
    #pragma unroll
    for (int t = 0; t < 4; ++t) {
      v8f c = (t == 0) ? c0 : (t == 1) ? c1 : (t == 2) ? c2 : c3;
      int col = colBase + t * 16 + nB;
      float bv = bias ? bias[col] : 0.0f;
      #pragma unroll
      for (int j = 0; j < 8; ++j) {
        int grow = m0 + hi16 * 8 + j;
        float v = c[j] + bv;
        float y = 0.5f * v * (1.0f + erff(v * 0.70710678118654752f));
        outB[(size_t)grow * OUTW + col] = (bf16_t)y;
      }
    }
    return;
  }

  // ------- LayerNorm epilogue (OUTW == 512, blockIdx.y == 0) -------
  #pragma unroll
  for (int t = 0; t < 4; ++t) {
    v8f c = (t == 0) ? c0 : (t == 1) ? c1 : (t == 2) ? c2 : c3;
    int col = wave * 64 + t * 16 + nB;
    float bv = bias ? bias[col] : 0.0f;
    #pragma unroll
    for (int j = 0; j < 8; ++j) {
      int r = hi16 * 8 + j;
      float v = c[j] + bv;
      if (addend) v += addend[(size_t)(m0 + r) * DIMD + col];
      Ct[r][col] = v;
    }
  }
  __syncthreads();

  #pragma unroll
  for (int rr = 0; rr < 2; ++rr) {
    int r = wave * 2 + rr;
    float sum = 0.0f, ssq = 0.0f;
    #pragma unroll
    for (int i = 0; i < DIMD / 32; ++i) {
      float v = Ct[r][lane + 32 * i];
      sum += v; ssq += v * v;
    }
    #pragma unroll
    for (int off = 16; off > 0; off >>= 1) {
      sum += __shfl_xor(sum, off, 32);
      ssq += __shfl_xor(ssq, off, 32);
    }
    float mu   = sum * (1.0f / DIMD);
    float var  = ssq * (1.0f / DIMD) - mu * mu;
    float rstd = rsqrtf(var + 1e-5f);
    #pragma unroll
    for (int i = 0; i < DIMD / 32; ++i) {
      int col = lane + 32 * i;
      float y = (Ct[r][col] - mu) * rstd * lnw[col] + lnb[col];
      size_t o = (size_t)(m0 + r) * DIMD + col;
      outF[o] = y;
      if (outB) outB[o] = (bf16_t)y;
    }
  }
}

// ================================================================ launch
extern "C" void kernel_launch(void* const* d_in, const int* in_sizes, int n_in,
                              void* d_out, int out_size, void* d_ws, size_t ws_size,
                              hipStream_t stream) {
  (void)in_sizes; (void)n_in; (void)out_size; (void)ws_size;

  const float* x      = (const float*)d_in[0];
  const int*   e_idx  = (const int*)  d_in[1];
  const int*   e_type = (const int*)  d_in[2];
  const float* self_W = (const float*)d_in[3];
  const float* self_b = (const float*)d_in[4];
  const float* rel_W  = (const float*)d_in[5];
  const float* rel_b  = (const float*)d_in[6];
  const float* ln1_w  = (const float*)d_in[7];
  const float* ln1_b  = (const float*)d_in[8];
  const float* W1     = (const float*)d_in[9];
  const float* b1     = (const float*)d_in[10];
  const float* W2     = (const float*)d_in[11];
  const float* b2     = (const float*)d_in[12];
  const float* ln2_w  = (const float*)d_in[13];
  const float* ln2_b  = (const float*)d_in[14];
  float* out = (float*)d_out;

  const int KCAT = DIMD * (RR + 1);  // 2560

  // ---- workspace layout (all offsets 256B aligned) ----
  char* w = (char*)d_ws;
  size_t off = 0;
  auto take = [&](size_t bytes) { size_t o = off; off += (bytes + 255) & ~(size_t)255; return o; };
  size_t s_off    = take((size_t)NN * RR * DIMD * 4);   // 81,920,000
  size_t cnt_off  = take((size_t)NN * RR * 4);          // 160,000
  size_t acat_off = take((size_t)NN * KCAT * 2);        // 51,200,000
  size_t a0_off   = take((size_t)NN * DIMD * 4);        // 20,480,000
  size_t wcat_off = take((size_t)KCAT * DIMD * 2);      // 2,621,440
  size_t w1t_off  = take((size_t)DIMD * 2 * DIMD * 2);  // 1,048,576
  size_t w2t_off  = take((size_t)2 * DIMD * DIMD * 2);  // 1,048,576

  float*  s_buf  = (float*)(w + s_off);
  float*  cnt    = (float*)(w + cnt_off);
  bf16_t* Acat   = (bf16_t*)(w + acat_off);
  float*  A0     = (float*)(w + a0_off);
  bf16_t* WcatT  = (bf16_t*)(w + wcat_off);
  bf16_t* W1t    = (bf16_t*)(w + w1t_off);
  bf16_t* W2t    = (bf16_t*)(w + w2t_off);
  // s region is dead after k_finalize -> reuse for x1 / gelu intermediates
  float*  x1f    = (float*)(w + s_off);                               // 20,480,000
  bf16_t* x1b    = (bf16_t*)(w + s_off + (size_t)NN * DIMD * 4);      // 10,240,000
  bf16_t* hgelu  = (bf16_t*)(w + s_off + (size_t)NN * DIMD * 6);      // 20,480,000

  // 1) zero the scatter accumulators (s and cnt are contiguous)
  k_zero<<<8192, 256, 0, stream>>>(s_buf, (size_t)NN * RR * DIMD + (size_t)NN * RR);

  // 2) convert+transpose weights to bf16 [OUTW][K]
  //    WcatT rows f: cols [0,512) = self_W[.,f], cols [512,2560) = rel_W flat[.,f]
  k_transpose_bf16<<<2048, 256, 0, stream>>>(self_W, WcatT, DIMD,     DIMD,     KCAT, 0);
  k_transpose_bf16<<<2048, 256, 0, stream>>>(rel_W,  WcatT, RR*DIMD,  DIMD,     KCAT, DIMD);
  k_transpose_bf16<<<2048, 256, 0, stream>>>(W1,     W1t,   DIMD,     2*DIMD,   DIMD, 0);
  k_transpose_bf16<<<2048, 256, 0, stream>>>(W2,     W2t,   2*DIMD,   DIMD,     2*DIMD, 0);

  // 3) scatter x rows into per-(dst,rel) accumulators (one wave per edge)
  k_edge_scatter<<<EE / 8, 256, 0, stream>>>(e_idx, e_type, x, s_buf, cnt);

  // 4) build A0 addend and bf16 concat activations [x | s*invdeg]
  k_finalize<<<NN, 256, 0, stream>>>(x, self_b, rel_b, s_buf, cnt, Acat, A0);

  // 5) GEMM1 (K=2560): self+relation projections, + A0, fused LN1 -> x1 (f32 + bf16)
  k_gemm_wmma<<<dim3(NN / 16, 1), 256, 0, stream>>>(
      Acat, WcatT, KCAT, DIMD, nullptr, A0, ln1_w, ln1_b, x1f, x1b, 0);

  // 6) GEMM2 (K=512, OUTW=1024): x1 @ W1 + b1, exact GELU -> hgelu (bf16)
  k_gemm_wmma<<<dim3(NN / 16, 2), 256, 0, stream>>>(
      x1b, W1t, DIMD, 2 * DIMD, b1, nullptr, nullptr, nullptr, nullptr, hgelu, 1);

  // 7) GEMM3 (K=1024): hgelu @ W2 + b2 + x1 residual, fused LN2 -> out (f32)
  k_gemm_wmma<<<dim3(NN / 16, 1), 256, 0, stream>>>(
      hgelu, W2t, 2 * DIMD, DIMD, b2, x1f, ln2_w, ln2_b, out, nullptr, 0);
}
